// EGNN_Model_44220983280256
// MI455X (gfx1250) — matrix operand order repaired and checked by reading, not compiled, not run on hardware
//
#include <hip/hip_runtime.h>

typedef _Float16 v16h __attribute__((ext_vector_type(16)));
typedef _Float16 v8h  __attribute__((ext_vector_type(8)));
typedef float    v8f  __attribute__((ext_vector_type(8)));

constexpr int NB = 16;      // batch
constexpr int NL = 1024;    // seq length
constexpr int ND = 16;      // emb dim
constexpr int NK = 8;       // neighbors
constexpr int FRAG = 512;   // halves per packed B fragment (32 lanes x 16 halves)

// Fast SiLU: x * v_rcp_f32(1+exp(-x)) instead of the IEEE div_scale/div_fmas
// sequence (divisor is in (1, inf), well conditioned; matches f16 data class).
__device__ __forceinline__ float fast_silu(float x) {
  return x * __builtin_amdgcn_rcpf(1.f + __expf(-x));
}

// ---------------------------------------------------------------------------
// Weight packing: B-fragment layout for v_wmma_f32_16x16x32_f16 (wave32).
// frag = tile*kchunks + chunk; lane n = tile*16 + (lane&15); h = lane>>4;
// element e -> K = chunk*32 + (e&7) + 8*h + 16*(e>>3)  (matches A fragment order).
// ---------------------------------------------------------------------------
__global__ void pack_wk(const float* __restrict__ src, _Float16* __restrict__ dst,
                        int fi, int fo, int kchunks) {
  int frag = blockIdx.x;
  int lane = threadIdx.x;            // 0..31
  int t = frag / kchunks, c = frag % kchunks;
  int n = t * 16 + (lane & 15);
  int h = lane >> 4;
#pragma unroll
  for (int e = 0; e < 16; ++e) {
    int k = c * 32 + (e & 7) + 8 * h + 16 * (e >> 3);
    float v = (k < fi && n < fo) ? src[(size_t)k * fo + n] : 0.f;
    dst[(size_t)frag * FRAG + lane * 16 + e] = (_Float16)v;
  }
}

// ---------------------------------------------------------------------------
// A-fragment loader from an LDS tile X[16][stride] (f16, row-major).
// 16-bit A layout: lane m=lane&15, h=lane>>4; element e -> K =
// chunk*32 + (e&7) + 8*h + 16*(e>>3)  => two contiguous 8-half (16B) groups.
// ---------------------------------------------------------------------------
__device__ __forceinline__ v16h ld_afrag(const _Float16* X, int stride, int chunk) {
  int lane = threadIdx.x & 31;
  int row = lane & 15, h = lane >> 4;
  const _Float16* p = X + row * stride + chunk * 32 + h * 8;
  v8h g0 = *(const v8h*)p;
  v8h g1 = *(const v8h*)(p + 16);
  v16h a;
#pragma unroll
  for (int i = 0; i < 8; ++i) { a[i] = g0[i]; a[i + 8] = g1[i]; }
  return a;
}

template <int KC>
__device__ __forceinline__ v8f wmma_tile(const _Float16* A, int strideA,
                                         const _Float16* __restrict__ Bpack, int tile) {
  int lane = threadIdx.x & 31;
  v8f acc = {};
#pragma unroll
  for (int c = 0; c < KC; ++c) {
    v16h a = ld_afrag(A, strideA, c);
    v16h b = *(const v16h*)(Bpack + (size_t)(tile * KC + c) * FRAG + (size_t)lane * 16);
    acc = __builtin_amdgcn_wmma_f32_16x16x32_f16(false, a, false, b, (short)0, acc,
                                                 false, false);
  }
  return acc;
}

// D layout: lane l (col n=l&15, half h=l>>4), vgpr r -> row m = r + 8*h.
__device__ __forceinline__ void store_tile_act(_Float16* O, int strideO, int tile, v8f acc,
                                               const float* __restrict__ bias, int realN,
                                               bool act) {
  int lane = threadIdx.x & 31;
  int n = lane & 15, h = lane >> 4;
  int ng = tile * 16 + n;
  float bv = (ng < realN) ? bias[ng] : 0.f;
#pragma unroll
  for (int r = 0; r < 8; ++r) {
    float x = acc[r] + bv;
    if (act) x = fast_silu(x);
    O[(r + 8 * h) * strideO + ng] = (_Float16)x;
  }
}

__device__ __forceinline__ unsigned long long shfl_xor_u64(unsigned long long v, int m) {
  int lo = __shfl_xor((int)(unsigned)v, m, 32);
  int hi = __shfl_xor((int)(v >> 32), m, 32);
  return ((unsigned long long)(unsigned)hi << 32) | (unsigned)lo;
}

// ---------------------------------------------------------------------------
// feats0 = token_emb[residues] + pos_emb ; coords copy
// ---------------------------------------------------------------------------
__global__ __launch_bounds__(256) void init_kernel(const float* __restrict__ coords,
                                                   const int* __restrict__ residues,
                                                   const float* __restrict__ tok,
                                                   const float* __restrict__ pos,
                                                   float* __restrict__ f0,
                                                   float* __restrict__ c0) {
  int n = blockIdx.x * 256 + threadIdx.x;  // 0..NB*NL-1
  int l = n & (NL - 1);
  int r = residues[n];
#pragma unroll
  for (int c = 0; c < ND; ++c)
    f0[(size_t)n * ND + c] = tok[r * ND + c] + pos[l * ND + c];
  c0[n * 3 + 0] = coords[n * 3 + 0];
  c0[n * 3 + 1] = coords[n * 3 + 1];
  c0[n * 3 + 2] = coords[n * 3 + 2];
}

// ---------------------------------------------------------------------------
// kNN: one wave per row; batch coords staged in LDS; per-lane sorted top-8
// then 8 rounds of wave-wide min-extraction.
// ---------------------------------------------------------------------------
__global__ __launch_bounds__(256) void knn_kernel(const float* __restrict__ cb,
                                                  const int* __restrict__ lengths,
                                                  int* __restrict__ knn) {
  __shared__ float sc[NL * 3];  // 12 KB: this batch's coords
  int tid = threadIdx.x;
  int gw0 = blockIdx.x * 8;                 // 8 rows per block, same batch
  int b = gw0 >> 10;
  size_t cb0 = (size_t)b * NL * 3;
  for (int i = tid; i < NL * 3; i += 256) sc[i] = cb[cb0 + i];
  __syncthreads();

  int wv = tid >> 5, lane = tid & 31;
  int l = (gw0 + wv) & (NL - 1);
  int len = lengths[b];
  bool maski = l < len;
  float xi = sc[l * 3 + 0], yi = sc[l * 3 + 1], zi = sc[l * 3 + 2];

  unsigned long long best[8];
#pragma unroll
  for (int i = 0; i < 8; ++i) best[i] = ~0ull;

  for (int t = 0; t < NL / 32; ++t) {
    int c = t * 32 + lane;
    float dx = xi - sc[c * 3 + 0];
    float dy = yi - sc[c * 3 + 1];
    float dz = zi - sc[c * 3 + 2];
    float d2 = dx * dx + dy * dy + dz * dz;
    bool pm = maski && (c < len);
    float r = pm ? d2 : 1e5f;
    if (pm && (c == l + 1 || c == l - 1)) r = 0.f;  // chain adjacency forced in
    if (c == l) r = -1.f;                           // self always first
    unsigned u = __float_as_uint(r);
    u = ((int)u < 0) ? ~u : (u | 0x80000000u);      // sortable float
    unsigned long long key = ((unsigned long long)u << 10) | (unsigned)c;
    if (key < best[7]) {
      best[7] = key;
#pragma unroll
      for (int i = 7; i > 0; --i)
        if (best[i] < best[i - 1]) {
          unsigned long long tmp = best[i]; best[i] = best[i - 1]; best[i - 1] = tmp;
        }
    }
  }

  size_t orow = (size_t)(gw0 + wv) * NK;
#pragma unroll
  for (int r8 = 0; r8 < 8; ++r8) {
    unsigned long long mv = best[0];
#pragma unroll
    for (int ofs = 16; ofs > 0; ofs >>= 1) {
      unsigned long long o = shfl_xor_u64(mv, ofs);
      if (o < mv) mv = o;
    }
    if (best[0] == mv) {  // pop winner (keys unique: idx in low bits)
#pragma unroll
      for (int i = 0; i < 7; ++i) best[i] = best[i + 1];
      best[7] = ~0ull;
    }
    if (lane == 0) knn[orow + r8] = (int)(mv & 1023ull);
  }
}

// ---------------------------------------------------------------------------
// Edge phase: one wave = 2 nodes = 16 edges = one WMMA M-tile.
// edge MLP (33->66->16), m_i reduction, coors MLP (16->64->1), coord update.
// ---------------------------------------------------------------------------
__global__ __launch_bounds__(256) void edge_kernel(
    const float* __restrict__ cb, const float* __restrict__ fb,
    const int* __restrict__ lengths, const int* __restrict__ knn,
    const _Float16* __restrict__ wp_e1, const _Float16* __restrict__ wp_e2,
    const _Float16* __restrict__ wp_c1, const float* __restrict__ eb1,
    const float* __restrict__ eb2, const float* __restrict__ cbias1,
    const float* __restrict__ cw2, const float* __restrict__ cb2v,
    float* __restrict__ cnext, float* __restrict__ mi_out) {
  __shared__ _Float16 smem[8 * 2560];  // 40 KB: per-wave [16x64 | 16x96] slabs
  int tid = threadIdx.x, wv = tid >> 5, lane = tid & 31;
  _Float16* buf0 = smem + wv * 2560;         // X [16][64], later M [16][32]
  _Float16* buf1 = smem + wv * 2560 + 1024;  // H [16][96], later G [16][64]

  int gw = blockIdx.x * 8 + wv;  // 0 .. NB*NL/2-1
  int b = gw >> 9;               // 512 node-pairs per batch
  int n0 = (gw & 511) * 2;
  size_t gbase = (size_t)b * NL + n0;
  int len = lengths[b];

  for (int i = lane; i < 2560; i += 32) smem[wv * 2560 + i] = (_Float16)0.f;
  __syncthreads();

  bool em = false;
  float rx = 0, ry = 0, rz = 0, cix = 0, ciy = 0, ciz = 0;
  if (lane < 16) {
    int row = lane;
    int nl = n0 + (row >> 3);
    int kk = row & 7;
    size_t gn = (size_t)b * NL + nl;
    int j = knn[gn * NK + kk];
    size_t gj = (size_t)b * NL + j;
    em = (nl < len) && (j < len);
    cix = cb[gn * 3 + 0]; ciy = cb[gn * 3 + 1]; ciz = cb[gn * 3 + 2];
    rx = cix - cb[gj * 3 + 0]; ry = ciy - cb[gj * 3 + 1]; rz = ciz - cb[gj * 3 + 2];
    float rd = rx * rx + ry * ry + rz * rz;
    const float* fi = fb + gn * ND;
    const float* fj = fb + gj * ND;
#pragma unroll
    for (int c = 0; c < ND; ++c) {
      buf0[row * 64 + c] = (_Float16)fi[c];
      buf0[row * 64 + 16 + c] = (_Float16)fj[c];
    }
    buf0[row * 64 + 32] = (_Float16)rd;  // cols 33..63 stay zero
  }
  unsigned embits = (unsigned)__ballot(em ? 1 : 0);  // rows 0..15 valid-edge bits
  __syncthreads();

  // GEMM1: [16x64(pad33)] @ edge_w1 -> H [16x66(pad80)], SiLU
#pragma unroll
  for (int t = 0; t < 5; ++t) {
    v8f acc = wmma_tile<2>(buf0, 64, wp_e1, t);
    store_tile_act(buf1, 96, t, acc, eb1, 66, true);  // cols 80..95 stay zero
  }
  __syncthreads();

  // GEMM2: [16x96(pad66)] @ edge_w2 -> m_ij [16x16], SiLU; fused masked m_i
  {
    v8f acc = wmma_tile<3>(buf1, 96, wp_e2, 0);
    int nn = lane & 15, h = lane >> 4;
    float bv = eb2[nn];
    float misum = 0.f;
#pragma unroll
    for (int r = 0; r < 8; ++r) {
      float x = fast_silu(acc[r] + bv);
      int row = r + 8 * h;
      if ((embits >> row) & 1u) misum += x;
      buf0[row * 32 + nn] = (_Float16)x;
      buf0[row * 32 + 16 + nn] = (_Float16)0.f;  // K-pad for coors GEMM
    }
    mi_out[(gbase + h) * ND + nn] = misum;
  }
  __syncthreads();

  // GEMM3: m_ij [16x32(pad16)] @ coors_w1 -> G [16x64], SiLU
#pragma unroll
  for (int t = 0; t < 4; ++t) {
    v8f acc = wmma_tile<1>(buf0, 32, wp_c1, t);
    store_tile_act(buf1, 64, t, acc, cbias1, 64, true);
  }
  __syncthreads();

  // matvec G @ coors_w2 + b -> w[16 edges]; masked equivariant coord update
  if (lane < 16) {
    int row = lane;
    float s = cb2v[0];
    for (int k2 = 0; k2 < 64; ++k2) s += (float)buf1[row * 64 + k2] * cw2[k2];
    float w = em ? s : 0.f;
    float fx = w * rx, fy = w * ry, fz = w * rz;
#pragma unroll
    for (int ofs = 4; ofs > 0; ofs >>= 1) {  // reduce over the 8 edges of a node
      fx += __shfl_xor(fx, ofs);
      fy += __shfl_xor(fy, ofs);
      fz += __shfl_xor(fz, ofs);
    }
    if ((lane & 7) == 0) {
      size_t gn = gbase + (lane >> 3);
      cnext[gn * 3 + 0] = cix + fx;
      cnext[gn * 3 + 1] = ciy + fy;
      cnext[gn * 3 + 2] = ciz + fz;
    }
  }
}

// ---------------------------------------------------------------------------
// Node phase: one wave = 16 nodes. LayerNorm + concat m_i, MLP 32->32->16,
// residual add.
// ---------------------------------------------------------------------------
__global__ __launch_bounds__(256) void node_kernel(
    const float* __restrict__ fb, const float* __restrict__ mi,
    const _Float16* __restrict__ wp_n1, const _Float16* __restrict__ wp_n2,
    const float* __restrict__ nb1, const float* __restrict__ nb2,
    const float* __restrict__ lng, const float* __restrict__ lnb,
    float* __restrict__ fnext) {
  __shared__ _Float16 smem[8 * 1024];  // per-wave [16x32 | 16x32]
  int tid = threadIdx.x, wv = tid >> 5, lane = tid & 31;
  _Float16* nbuf = smem + wv * 1024;
  _Float16* hbuf = smem + wv * 1024 + 512;
  size_t base = ((size_t)blockIdx.x * 8 + wv) * 16;  // global node base

  if (lane < 16) {
    int row = lane;
    size_t gn = base + row;
    float f[16], mu = 0.f;
#pragma unroll
    for (int c = 0; c < ND; ++c) { f[c] = fb[gn * ND + c]; mu += f[c]; }
    mu *= (1.f / 16.f);
    float var = 0.f;
#pragma unroll
    for (int c = 0; c < ND; ++c) { float dd = f[c] - mu; var += dd * dd; }
    float inv = rsqrtf(var * (1.f / 16.f) + 1e-5f);
#pragma unroll
    for (int c = 0; c < ND; ++c) {
      nbuf[row * 32 + c] = (_Float16)((f[c] - mu) * inv * lng[c] + lnb[c]);
      nbuf[row * 32 + 16 + c] = (_Float16)mi[gn * ND + c];
    }
  }
  __syncthreads();
#pragma unroll
  for (int t = 0; t < 2; ++t) {
    v8f acc = wmma_tile<1>(nbuf, 32, wp_n1, t);
    store_tile_act(hbuf, 32, t, acc, nb1, 32, true);
  }
  __syncthreads();
  {
    v8f acc = wmma_tile<1>(hbuf, 32, wp_n2, 0);
    int nn = lane & 15, h = lane >> 4;
    float bv = nb2[nn];
#pragma unroll
    for (int r = 0; r < 8; ++r) {
      size_t gn = base + r + 8 * h;
      fnext[gn * ND + nn] = fb[gn * ND + nn] + (acc[r] + bv);  // residual
    }
  }
}

// ---------------------------------------------------------------------------
// Final projection: delta = feats @ final_w + final_b
// ---------------------------------------------------------------------------
__global__ __launch_bounds__(256) void final_kernel(const float* __restrict__ fb,
                                                    const float* __restrict__ fw,
                                                    const float* __restrict__ fbias,
                                                    float* __restrict__ out) {
  int n = blockIdx.x * 256 + threadIdx.x;  // 0..NB*NL-1
  float f[16];
#pragma unroll
  for (int c = 0; c < ND; ++c) f[c] = fb[(size_t)n * ND + c];
#pragma unroll
  for (int o = 0; o < 3; ++o) {
    float s = fbias[o];
#pragma unroll
    for (int k = 0; k < ND; ++k) s += f[k] * fw[k * 3 + o];
    out[(size_t)n * 3 + o] = s;
  }
}

// ---------------------------------------------------------------------------
extern "C" void kernel_launch(void* const* d_in, const int* in_sizes, int n_in,
                              void* d_out, int out_size, void* d_ws, size_t ws_size,
                              hipStream_t stream) {
  (void)in_sizes; (void)n_in; (void)out_size; (void)ws_size;
  const float* coords   = (const float*)d_in[0];
  const int*   residues = (const int*)d_in[1];
  const int*   lengths  = (const int*)d_in[2];
  const float* tok      = (const float*)d_in[3];
  const float* pos      = (const float*)d_in[4];
  const float* ew1 = (const float*)d_in[5];
  const float* eb1 = (const float*)d_in[6];
  const float* ew2 = (const float*)d_in[7];
  const float* eb2 = (const float*)d_in[8];
  const float* cw1 = (const float*)d_in[9];
  const float* cb1 = (const float*)d_in[10];
  const float* cw2 = (const float*)d_in[11];
  const float* cb2 = (const float*)d_in[12];
  const float* lng = (const float*)d_in[13];
  const float* lnb = (const float*)d_in[14];
  const float* nw1 = (const float*)d_in[15];
  const float* nb1 = (const float*)d_in[16];
  const float* nw2 = (const float*)d_in[17];
  const float* nb2 = (const float*)d_in[18];
  const float* fw  = (const float*)d_in[19];
  const float* fbv = (const float*)d_in[20];

  char* ws = (char*)d_ws;
  size_t off = 0;
  auto carve = [&](size_t bytes) -> char* {
    char* p = ws + off;
    off += (bytes + 255) & ~(size_t)255;
    return p;
  };
  float* fbuf0 = (float*)carve(sizeof(float) * NB * NL * ND);
  float* fbuf1 = (float*)carve(sizeof(float) * NB * NL * ND);
  float* cbuf0 = (float*)carve(sizeof(float) * NB * NL * 3);
  float* cbuf1 = (float*)carve(sizeof(float) * NB * NL * 3);
  float* mibuf = (float*)carve(sizeof(float) * NB * NL * ND);
  int*   knnb  = (int*)carve(sizeof(int) * NB * NL * NK);
  _Float16* wpack = (_Float16*)carve(sizeof(_Float16) * 3 * 20 * FRAG);

  // Pack all weights into WMMA B-fragment layout (20 fragments per depth).
  for (int d = 0; d < 3; ++d) {
    _Float16* base = wpack + (size_t)d * 20 * FRAG;
    pack_wk<<<10, 32, 0, stream>>>(ew1 + d * 33 * 66, base,             33, 66, 2);
    pack_wk<<<3,  32, 0, stream>>>(ew2 + d * 66 * 16, base + 10 * FRAG, 66, 16, 3);
    pack_wk<<<4,  32, 0, stream>>>(cw1 + d * 16 * 64, base + 13 * FRAG, 16, 64, 1);
    pack_wk<<<2,  32, 0, stream>>>(nw1 + d * 32 * 32, base + 17 * FRAG, 32, 32, 1);
    pack_wk<<<1,  32, 0, stream>>>(nw2 + d * 32 * 16, base + 19 * FRAG, 32, 16, 1);
  }

  init_kernel<<<NB * NL / 256, 256, 0, stream>>>(coords, residues, tok, pos, fbuf0, cbuf0);

  float* fcur = fbuf0; float* fnxt = fbuf1;
  float* ccur = cbuf0; float* cnxt = cbuf1;
  for (int d = 0; d < 3; ++d) {
    _Float16* base = wpack + (size_t)d * 20 * FRAG;
    knn_kernel<<<NB * NL / 8, 256, 0, stream>>>(ccur, lengths, knnb);
    edge_kernel<<<NB * NL / 16, 256, 0, stream>>>(
        ccur, fcur, lengths, knnb, base, base + 10 * FRAG, base + 13 * FRAG,
        eb1 + d * 66, eb2 + d * 16, cb1 + d * 64, cw2 + d * 64, cb2 + d,
        cnxt, mibuf);
    node_kernel<<<NB * NL / 128, 256, 0, stream>>>(
        fcur, mibuf, base + 17 * FRAG, base + 19 * FRAG,
        nb1 + d * 32, nb2 + d * 16, lng + d * 16, lnb + d * 16, fnxt);
    float* t = fcur; fcur = fnxt; fnxt = t;
    t = ccur; ccur = cnxt; cnxt = t;
  }

  final_kernel<<<NB * NL / 256, 256, 0, stream>>>(fcur, fw, fbv, (float*)d_out);
}
